// EfficientMultiHeadAttention_43825846288421
// MI455X (gfx1250) — matrix operand order repaired and checked by reading, not compiled, not run on hardware
//
#include <hip/hip_runtime.h>

typedef _Float16 f16;
typedef __attribute__((ext_vector_type(8)))  _Float16 v8h;
typedef __attribute__((ext_vector_type(16))) _Float16 v16h;
typedef __attribute__((ext_vector_type(8)))  float    v8f;

// ---------------------------------------------------------------------------
// helpers
// ---------------------------------------------------------------------------
__device__ __forceinline__ v16h join16(v8h lo, v8h hi) {
  return __builtin_shufflevector(lo, hi, 0,1,2,3,4,5,6,7,8,9,10,11,12,13,14,15);
}
__device__ __forceinline__ v8f wmma_f16(v16h a, v16h b, v8f c) {
  // D = A(16x32 f16) * B(32x16 f16) + C(16x16 f32)
  return __builtin_amdgcn_wmma_f32_16x16x32_f16(false, a, false, b, (short)0, c,
                                                false, false);
}
__device__ __forceinline__ v8h cvt8(const float* p) {
  float4 a = *(const float4*)p;
  float4 b = *(const float4*)(p + 4);
  v8h r;
  r[0]=(f16)a.x; r[1]=(f16)a.y; r[2]=(f16)a.z; r[3]=(f16)a.w;
  r[4]=(f16)b.x; r[5]=(f16)b.y; r[6]=(f16)b.z; r[7]=(f16)b.w;
  return r;
}
// CDNA5 async memory->LDS copy, 16 bytes per lane, tracked by ASYNCcnt.
// LDS flat aperture keeps the byte offset in addr[31:0], so truncating the
// generic pointer yields the LDS address the instruction needs.
__device__ __forceinline__ void async_cp16(void* lds, const void* gaddr) {
  unsigned l = (unsigned)(uintptr_t)lds;
  unsigned long long g = (unsigned long long)(uintptr_t)gaddr;
  asm volatile("global_load_async_to_lds_b128 %0, %1, off"
               :: "v"(l), "v"(g) : "memory");
}
__device__ __forceinline__ void wait_async() {
  asm volatile("s_wait_asynccnt 0" ::: "memory");
}

// problem constants
#define NB   4
#define SQ   2048
#define DM   1024
#define NH   16
#define HD   64

// ---------------------------------------------------------------------------
// Kernel 1: qkv = hidden(8192x1024,f32) @ w_qkv(1024x3072,f32) -> q,k,v f16
//   block tile 128(M) x 128(N), K-step 32; 8 waves of 32x64 (8 wmma/kstep)
//   q is pre-scaled by 1/sqrt(hd) = 0.125
// ---------------------------------------------------------------------------
__global__ __launch_bounds__(256) void qkv_kernel(
    const float* __restrict__ A, const float* __restrict__ B,
    f16* __restrict__ qb, f16* __restrict__ kb, f16* __restrict__ vb)
{
  const int K = DM, N = 3 * DM;
  __shared__ __align__(16) f16 As[128 * 40];  // [row 0..127][k 0..31]
  __shared__ __align__(16) f16 Bs[128 * 40];  // [n 0..127][k 0..31] (transposed)

  const int tid  = threadIdx.x;
  const int wid  = tid >> 5, lane = tid & 31;
  const int lh   = lane >> 4, lm = lane & 15;
  const int bm   = blockIdx.x * 128, bn = blockIdx.y * 128;
  const int wrow = (wid >> 1) * 32, wcol = (wid & 1) * 64;

  const int arow = tid >> 1, ac0 = (tid & 1) * 16;   // A staging: 128x32, 16/thr
  const int brow = tid >> 3, bn0 = (tid & 7) * 16;   // B staging: 32x128, 16/thr

  v8f acc[2][4] = {};

  for (int kk = 0; kk < K; kk += 32) {
    // ---- stage A tile (f32 -> f16)
    {
      const float* ap = A + (size_t)(bm + arow) * K + kk + ac0;
      *(v8h*)&As[arow * 40 + ac0]     = cvt8(ap);
      *(v8h*)&As[arow * 40 + ac0 + 8] = cvt8(ap + 8);
    }
    // ---- stage B tile transposed (f32 -> f16), Bs[n][k]
    {
      const float* bp = B + (size_t)(kk + brow) * N + bn + bn0;
      float4 f0 = *(const float4*)bp;
      float4 f1 = *(const float4*)(bp + 4);
      float4 f2 = *(const float4*)(bp + 8);
      float4 f3 = *(const float4*)(bp + 12);
      float vals[16] = {f0.x,f0.y,f0.z,f0.w, f1.x,f1.y,f1.z,f1.w,
                        f2.x,f2.y,f2.z,f2.w, f3.x,f3.y,f3.z,f3.w};
      #pragma unroll
      for (int i = 0; i < 16; i++) Bs[(bn0 + i) * 40 + brow] = (f16)vals[i];
    }
    __syncthreads();

    // ---- fragments (preloaded) + 8 WMMAs
    const f16* a0p = &As[(wrow + lm) * 40 + lh * 8];
    const f16* a1p = &As[(wrow + 16 + lm) * 40 + lh * 8];
    v16h a0 = join16(*(const v8h*)a0p, *(const v8h*)(a0p + 16));
    v16h a1 = join16(*(const v8h*)a1p, *(const v8h*)(a1p + 16));
    v16h b[4];
    #pragma unroll
    for (int t = 0; t < 4; t++) {
      const f16* bfp = &Bs[(wcol + t * 16 + lm) * 40 + lh * 8];
      b[t] = join16(*(const v8h*)bfp, *(const v8h*)(bfp + 16));
    }
    #pragma unroll
    for (int t = 0; t < 4; t++) {
      acc[0][t] = wmma_f16(a0, b[t], acc[0][t]);
      acc[1][t] = wmma_f16(a1, b[t], acc[1][t]);
    }
    __syncthreads();
  }

  // ---- epilogue: block-uniform routing to q (scaled) / k / v, [b][h][s][hd]
  const int part = bn >> 10;                       // 0:q 1:k 2:v (uniform)
  f16* dst = (part == 0) ? qb : ((part == 1) ? kb : vb);
  const float scale = (part == 0) ? 0.125f : 1.0f;
  #pragma unroll
  for (int r = 0; r < 2; r++) {
    #pragma unroll
    for (int t = 0; t < 4; t++) {
      #pragma unroll
      for (int j = 0; j < 8; j++) {
        int m = bm + wrow + r * 16 + j + 8 * lh;
        int n = bn + wcol + t * 16 + lm;
        int bidx = m >> 11, s = m & (SQ - 1);
        int rem = n & (DM - 1);
        int h = rem >> 6, d = rem & (HD - 1);
        size_t idx = (((size_t)bidx * NH + h) * SQ + s) * HD + d;
        dst[idx] = (f16)(acc[r][t][j] * scale);
      }
    }
  }
}

// ---------------------------------------------------------------------------
// Kernel 2: flash attention per (b,h): ctx = softmax(q k^T / 8 + mask) v
//   block = (b,h, 128 q rows), 8 waves each own 16 q rows; key blocks of 64
//   k tile staged with async global->LDS copies (ASYNCcnt)
// ---------------------------------------------------------------------------
__global__ __launch_bounds__(256) void attn_kernel(
    const f16* __restrict__ q, const f16* __restrict__ k,
    const f16* __restrict__ v, const int* __restrict__ mask,
    f16* __restrict__ ctx)
{
  __shared__ __align__(16) f16 kt[64 * 72];      // [key][hd]   stride 72
  __shared__ __align__(16) f16 vt[64 * 72];      // [hd][key]   stride 72
  __shared__ __align__(16) f16 pt[8 * 16 * 72];  // per-wave 16x64, stride 72

  const int tid = threadIdx.x;
  const int wid = tid >> 5, lane = tid & 31;
  const int lh  = lane >> 4, lm = lane & 15;
  const int bh  = blockIdx.x >> 4;               // 0..63 (= b*16+h)
  const int q0  = ((blockIdx.x & 15) << 7) + wid * 16;
  const int bidx = bh >> 4;

  const f16* qbase = q + ((size_t)bh * SQ + q0) * HD;
  const f16* kbase = k + (size_t)bh * SQ * HD;
  const f16* vbase = v + (size_t)bh * SQ * HD;
  const int* mbase = mask + bidx * SQ;

  // q A-fragments (K = hd): loaded once, scale already folded in
  const f16* qp = qbase + (size_t)lm * HD;
  v16h qa0 = join16(*(const v8h*)(qp + lh * 16),      *(const v8h*)(qp + 8 + lh * 16));
  v16h qa1 = join16(*(const v8h*)(qp + 32 + lh * 16), *(const v8h*)(qp + 40 + lh * 16));

  v8f acc[4] = {};
  float mrow[8], lrow[8];
  #pragma unroll
  for (int j = 0; j < 8; j++) { mrow[j] = -1e30f; lrow[j] = 0.f; }

  const int skey = tid >> 2, sd0 = (tid & 3) * 16;  // staging: 16 f16 / thread

  for (int kbk = 0; kbk < SQ; kbk += 64) {
    // ---- k block [key][hd]: async global->LDS, 2 x 16B per lane
    {
      const f16* kp = kbase + (size_t)(kbk + skey) * HD + sd0;
      async_cp16(&kt[skey * 72 + sd0],     kp);
      async_cp16(&kt[skey * 72 + sd0 + 8], kp + 8);
      // v block transposed [hd][key]: VGPR round-trip (transpose)
      const f16* vp = vbase + (size_t)(kbk + skey) * HD + sd0;
      v8h v0 = *(const v8h*)vp, v1 = *(const v8h*)(vp + 8);
      #pragma unroll
      for (int i = 0; i < 8; i++) {
        vt[(sd0 + i) * 72 + skey]     = v0[i];
        vt[(sd0 + 8 + i) * 72 + skey] = v1[i];
      }
    }
    wait_async();
    __syncthreads();

    // ---- scores: 4 N-tiles of 16 keys, K = hd = 64 (2 wmma each)
    v8f sv[4];
    #pragma unroll
    for (int n = 0; n < 4; n++) {
      const f16* bp = &kt[(n * 16 + lm) * 72 + lh * 16];
      v16h b0 = join16(*(const v8h*)bp,        *(const v8h*)(bp + 8));
      v16h b1 = join16(*(const v8h*)(bp + 32), *(const v8h*)(bp + 40));
      v8f sc = {};
      sc = wmma_f16(qa0, b0, sc);
      sc = wmma_f16(qa1, b1, sc);
      sv[n] = sc;
    }

    // ---- mask + online softmax (rows live at M = j + 8*lh, col = lm)
    int mv[4];
    #pragma unroll
    for (int n = 0; n < 4; n++) mv[n] = mbase[kbk + n * 16 + lm];
    float s[4][8];
    #pragma unroll
    for (int n = 0; n < 4; n++)
      #pragma unroll
      for (int j = 0; j < 8; j++)
        s[n][j] = (mv[n] != 0) ? sv[n][j] : -10000.0f;

    #pragma unroll
    for (int j = 0; j < 8; j++) {
      float tm = fmaxf(fmaxf(s[0][j], s[1][j]), fmaxf(s[2][j], s[3][j]));
      #pragma unroll
      for (int off = 8; off; off >>= 1) tm = fmaxf(tm, __shfl_xor(tm, off, 32));
      float mn   = fmaxf(mrow[j], tm);
      float corr = __expf(mrow[j] - mn);
      mrow[j] = mn;
      float ps = 0.f;
      #pragma unroll
      for (int n = 0; n < 4; n++) { float p = __expf(s[n][j] - mn); s[n][j] = p; ps += p; }
      #pragma unroll
      for (int off = 8; off; off >>= 1) ps += __shfl_xor(ps, off, 32);
      lrow[j] = lrow[j] * corr + ps;
      #pragma unroll
      for (int n = 0; n < 4; n++) acc[n][j] *= corr;
    }

    // ---- transpose P (C-layout -> A-layout) through per-wave LDS
    f16* pw = &pt[wid * 16 * 72];
    #pragma unroll
    for (int n = 0; n < 4; n++)
      #pragma unroll
      for (int j = 0; j < 8; j++)
        pw[(j + 8 * lh) * 72 + n * 16 + lm] = (f16)s[n][j];
    asm volatile("s_wait_dscnt 0" ::: "memory");

    const f16* pp = &pt[wid * 16 * 72 + lm * 72];
    v16h pa0 = join16(*(const v8h*)(pp + lh * 16),      *(const v8h*)(pp + 8 + lh * 16));
    v16h pa1 = join16(*(const v8h*)(pp + 32 + lh * 16), *(const v8h*)(pp + 40 + lh * 16));

    // ---- ctx += P(16x64) @ V(64x64): 4 hd-tiles, K = keys (2 wmma each)
    #pragma unroll
    for (int n = 0; n < 4; n++) {
      const f16* vp2 = &vt[(n * 16 + lm) * 72 + lh * 16];
      v16h b0 = join16(*(const v8h*)vp2,        *(const v8h*)(vp2 + 8));
      v16h b1 = join16(*(const v8h*)(vp2 + 32), *(const v8h*)(vp2 + 40));
      acc[n] = wmma_f16(pa0, b0, acc[n]);
      acc[n] = wmma_f16(pa1, b1, acc[n]);
    }
    __syncthreads();
  }

  // ---- epilogue: ctx[b][s][h*64+d] f16, normalized by row sum
  #pragma unroll
  for (int j = 0; j < 8; j++) {
    float inv = (lrow[j] > 0.f) ? 1.0f / lrow[j] : 0.f;
    int srow = q0 + j + 8 * lh;
    size_t base = ((size_t)bidx * SQ + srow) * DM + (bh & 15) * HD;
    #pragma unroll
    for (int n = 0; n < 4; n++)
      ctx[base + n * 16 + lm] = (f16)(acc[n][j] * inv);
  }
}

// ---------------------------------------------------------------------------
// Kernel 3: out = ctx(8192x1024,f16) @ w_out(1024x1024,f32) -> f32
//   A tile staged with async global->LDS copies (ASYNCcnt)
// ---------------------------------------------------------------------------
__global__ __launch_bounds__(256) void out_kernel(
    const f16* __restrict__ A, const float* __restrict__ B,
    float* __restrict__ C)
{
  const int K = DM, N = DM;
  __shared__ __align__(16) f16 As[128 * 40];
  __shared__ __align__(16) f16 Bs[128 * 40];

  const int tid  = threadIdx.x;
  const int wid  = tid >> 5, lane = tid & 31;
  const int lh   = lane >> 4, lm = lane & 15;
  const int bm   = blockIdx.x * 128, bn = blockIdx.y * 128;
  const int wrow = (wid >> 1) * 32, wcol = (wid & 1) * 64;

  const int arow = tid >> 1, ac0 = (tid & 1) * 16;
  const int brow = tid >> 3, bn0 = (tid & 7) * 16;

  v8f acc[2][4] = {};

  for (int kk = 0; kk < K; kk += 32) {
    // ---- stage A tile (f16, contiguous): async global->LDS
    {
      const f16* ap = A + (size_t)(bm + arow) * K + kk + ac0;
      async_cp16(&As[arow * 40 + ac0],     ap);
      async_cp16(&As[arow * 40 + ac0 + 8], ap + 8);
    }
    // ---- stage B tile transposed (f32 -> f16), Bs[n][k]
    {
      const float* bp = B + (size_t)(kk + brow) * N + bn + bn0;
      float4 f0 = *(const float4*)bp;
      float4 f1 = *(const float4*)(bp + 4);
      float4 f2 = *(const float4*)(bp + 8);
      float4 f3 = *(const float4*)(bp + 12);
      float vals[16] = {f0.x,f0.y,f0.z,f0.w, f1.x,f1.y,f1.z,f1.w,
                        f2.x,f2.y,f2.z,f2.w, f3.x,f3.y,f3.z,f3.w};
      #pragma unroll
      for (int i = 0; i < 16; i++) Bs[(bn0 + i) * 40 + brow] = (f16)vals[i];
    }
    wait_async();
    __syncthreads();

    const f16* a0p = &As[(wrow + lm) * 40 + lh * 8];
    const f16* a1p = &As[(wrow + 16 + lm) * 40 + lh * 8];
    v16h a0 = join16(*(const v8h*)a0p, *(const v8h*)(a0p + 16));
    v16h a1 = join16(*(const v8h*)a1p, *(const v8h*)(a1p + 16));
    v16h b[4];
    #pragma unroll
    for (int t = 0; t < 4; t++) {
      const f16* bfp = &Bs[(wcol + t * 16 + lm) * 40 + lh * 8];
      b[t] = join16(*(const v8h*)bfp, *(const v8h*)(bfp + 16));
    }
    #pragma unroll
    for (int t = 0; t < 4; t++) {
      acc[0][t] = wmma_f16(a0, b[t], acc[0][t]);
      acc[1][t] = wmma_f16(a1, b[t], acc[1][t]);
    }
    __syncthreads();
  }

  #pragma unroll
  for (int r = 0; r < 2; r++) {
    #pragma unroll
    for (int t = 0; t < 4; t++) {
      #pragma unroll
      for (int j = 0; j < 8; j++) {
        int m = bm + wrow + r * 16 + j + 8 * lh;
        int n = bn + wcol + t * 16 + lm;
        C[(size_t)m * N + n] = acc[r][t][j];
      }
    }
  }
}

// ---------------------------------------------------------------------------
// launcher
// ---------------------------------------------------------------------------
extern "C" void kernel_launch(void* const* d_in, const int* in_sizes, int n_in,
                              void* d_out, int out_size, void* d_ws, size_t ws_size,
                              hipStream_t stream) {
  const float* hs   = (const float*)d_in[0];  // (4,2048,1024) f32
  const int*   mask = (const int*)d_in[1];    // (4,2048) i32
  const float* wqkv = (const float*)d_in[2];  // (1024,3072) f32
  const float* wout = (const float*)d_in[3];  // (1024,1024) f32
  float* out = (float*)d_out;                 // (4,2048,1024) f32

  char* ws = (char*)d_ws;
  const size_t QKV_BYTES = (size_t)NB * NH * SQ * HD * sizeof(f16);  // 16 MB each
  f16* qb  = (f16*)(ws);
  f16* kb  = (f16*)(ws + QKV_BYTES);
  f16* vb  = (f16*)(ws + 2 * QKV_BYTES);
  f16* ctx = (f16*)(ws + 3 * QKV_BYTES);      // 16 MB

  dim3 blk(256);
  qkv_kernel<<<dim3((NB * SQ) / 128, (3 * DM) / 128), blk, 0, stream>>>(
      hs, wqkv, qb, kb, vb);
  attn_kernel<<<dim3(NB * NH * (SQ / 128)), blk, 0, stream>>>(
      qb, kb, vb, mask, ctx);
  out_kernel<<<dim3((NB * SQ) / 128, DM / 128), blk, 0, stream>>>(
      ctx, wout, out);
}
